// GATEncoder_29815662969290
// MI455X (gfx1250) — matrix operand order repaired and checked by reading, not compiled, not run on hardware
//
#include <hip/hip_runtime.h>
#include <hip/hip_bf16.h>

typedef __attribute__((ext_vector_type(16))) _Float16 v16h;
typedef __attribute__((ext_vector_type(8)))  float    v8f;

#define NNODES 10000
#define BB 4
#define DD 128
#define HH 4
#define DH 32

#define XS_LD 136
#define WS_LD 136

// ---------------------------------------------------------------------------
// GEMM: Y[M,128] = X · W^T (+bias).  X is [M, 128*chunks] given as up to two
// chunk pointers (X0 cols 0..127, X1 cols 128..255) so the MLP concat input
// never needs to be materialized.  W is [128, ldW] row-major (row o = output
// channel o).  f32 inputs staged into LDS as f16, accumulated in f32 via
// v_wmma_f32_16x16x32_f16.  Block = 256 thr (8 waves) -> 64 rows x 128 cols.
// Per k-step: preload A + all 4 B fragments, then issue the 4 independent
// WMMAs back-to-back so DS latency overlaps with WMMA execution instead of
// a per-WMMA s_wait_dscnt stall.
// ---------------------------------------------------------------------------
__launch_bounds__(256)
__global__ void gemm_xwT_wmma(const float* __restrict__ X0,
                              const float* __restrict__ X1,
                              const float* __restrict__ W,
                              const float* __restrict__ bias,
                              float* __restrict__ Y,
                              int M, int chunks, int ldW) {
  __shared__ _Float16 Xs[64 * XS_LD];
  __shared__ _Float16 Ws[128 * WS_LD];

  const int tid  = threadIdx.x;
  const int lane = tid & 31;
  const int wave = tid >> 5;
  const int row0 = blockIdx.x * 64;
  const int rt   = wave & 3;   // row tile (16 rows)
  const int cg   = wave >> 2;  // col group: 4 col tiles each

  v8f acc[4] = {};

  for (int kc = 0; kc < chunks; ++kc) {
    const float* Xsrc = (kc == 0) ? X0 : X1;
    // stage X tile: 64 rows x 128 k  (f32 -> f16)
    for (int i = tid; i < 64 * 128; i += 256) {
      int r = i >> 7, c = i & 127;
      int rr = row0 + r; if (rr >= M) rr = M - 1;
      Xs[r * XS_LD + c] = (_Float16)Xsrc[(size_t)rr * 128 + c];
    }
    // stage W slice: 128 out-channels x 128 k
    for (int i = tid; i < 128 * 128; i += 256) {
      int r = i >> 7, c = i & 127;
      Ws[r * WS_LD + c] = (_Float16)W[(size_t)r * ldW + kc * 128 + c];
    }
    __syncthreads();

    const int m   = lane & 15;
    const int kbA = (lane >> 4) * 8;    // A: K interleaved in groups of 8
    const int kbB = (lane >> 4) * 8;    // B: K interleaved in groups of 8
#pragma unroll
    for (int t = 0; t < 4; ++t) {       // K = 128 = 4 x 32
      // preload A fragment
      v16h a;
      const _Float16* xr = &Xs[(rt * 16 + m) * XS_LD + t * 32];
#pragma unroll
      for (int i = 0; i < 16; ++i)
        a[i] = xr[(i >> 3) * 16 + kbA + (i & 7)];
      // preload all 4 B fragments (independent DS loads, one wait)
      v16h bf[4];
#pragma unroll
      for (int c = 0; c < 4; ++c) {
        const _Float16* wr = &Ws[((cg * 4 + c) * 16 + m) * WS_LD + t * 32 + kbB];
#pragma unroll
        for (int i = 0; i < 16; ++i) bf[c][i] = wr[(i >> 3) * 16 + (i & 7)];
      }
      // 4 independent accumulator chains -> WMMAs issue back-to-back
#pragma unroll
      for (int c = 0; c < 4; ++c)
        acc[c] = __builtin_amdgcn_wmma_f32_16x16x32_f16(
            false, a, false, bf[c], (short)0, acc[c], false, false);
    }
    __syncthreads();
  }

  // D layout: VGPR v -> M = v (+8 for lanes 16-31), N = lane&15
  const int colb = lane & 15;
  const int radd = (lane >> 4) * 8;
#pragma unroll
  for (int c = 0; c < 4; ++c) {
    int col = (cg * 4 + c) * 16 + colb;
    float bv = bias ? bias[col] : 0.0f;
#pragma unroll
    for (int v = 0; v < 8; ++v) {
      int row = row0 + rt * 16 + v + radd;
      if (row < M) Y[(size_t)row * 128 + col] = acc[c][v] + bv;
    }
  }
}

// ---------------------------------------------------------------------------
// el/er: per (b,n,h) dot over DH
// ---------------------------------------------------------------------------
__global__ void lr_kernel(const float* __restrict__ feat,
                          const float* __restrict__ al,
                          const float* __restrict__ ar,
                          float* __restrict__ el, float* __restrict__ er,
                          int MH) {
  int idx = blockIdx.x * blockDim.x + threadIdx.x;
  if (idx >= MH) return;
  int bn = idx >> 2, h = idx & 3;
  const float* f = feat + (size_t)bn * DD + h * DH;
  const float* a = al + h * DH;
  const float* r = ar + h * DH;
  float sl = 0.f, sr = 0.f;
#pragma unroll
  for (int i = 0; i < DH; ++i) { sl += f[i] * a[i]; sr += f[i] * r[i]; }
  el[idx] = sl;
  er[idx] = sr;
}

// order-preserving float<->uint encoding for atomic max on signed floats
__device__ __forceinline__ unsigned enc_f(float f) {
  unsigned u = __float_as_uint(f);
  return (u & 0x80000000u) ? ~u : (u | 0x80000000u);
}
__device__ __forceinline__ float dec_f(unsigned u) {
  return (u & 0x80000000u) ? __uint_as_float(u & 0x7FFFFFFFu)
                           : __uint_as_float(~u);
}

__global__ void init_kernel(unsigned* __restrict__ menc,
                            float* __restrict__ den,
                            float* __restrict__ agg, int NBH) {
  int idx = blockIdx.x * blockDim.x + threadIdx.x;
  if (idx < NBH * DH) agg[idx] = 0.f;
  if (idx < NBH) { menc[idx] = enc_f(-__builtin_inff()); den[idx] = 0.f; }
}

// per (edge, b, h): score = leaky(el[src]+er[dst]) * w ; segment-max into dst
__global__ void edge_score(const int* __restrict__ src,
                           const int* __restrict__ dst,
                           const float* __restrict__ ew,
                           const float* __restrict__ el,
                           const float* __restrict__ er,
                           float* __restrict__ ebuf,
                           unsigned* __restrict__ menc, int E_, int N_) {
  int idx = blockIdx.x * blockDim.x + threadIdx.x;
  if (idx >= E_ * 16) return;
  int e = idx >> 4, bh = idx & 15;
  int b = bh >> 2, h = bh & 3;
  int s = src[e], d = dst[e];
  float v = el[(b * N_ + s) * HH + h] + er[(b * N_ + d) * HH + h];
  v = (v > 0.f) ? v : 0.1f * v;       // LeakyReLU(0.1)
  v *= ew[e];                          // edge weight pre-softmax
  ebuf[idx] = v;
  atomicMax(&menc[d * 16 + bh], enc_f(v));
}

// per (edge, b, h): ex = exp(e - m[dst]); segment-sum into den
__global__ void edge_exp(const int* __restrict__ dst,
                         float* __restrict__ ebuf,
                         const unsigned* __restrict__ menc,
                         float* __restrict__ den, int E_) {
  int idx = blockIdx.x * blockDim.x + threadIdx.x;
  if (idx >= E_ * 16) return;
  int e = idx >> 4, bh = idx & 15;
  int d = dst[e];
  float mv = dec_f(menc[d * 16 + bh]);
  if (!__builtin_isfinite(mv)) mv = 0.f;
  float ex = expf(ebuf[idx] - mv);
  ebuf[idx] = ex;
  atomicAdd(&den[d * 16 + bh], ex);
}

// one 512-thread block per edge: lanes = (b,h) x dh; gather feat[src],
// scale by attention, atomic-scatter into agg[dst]
__global__ void edge_agg(const int* __restrict__ src,
                         const int* __restrict__ dst,
                         const float* __restrict__ ebuf,
                         const float* __restrict__ den,
                         const float* __restrict__ feat,
                         float* __restrict__ agg, int N_) {
  int e  = blockIdx.x;
  int t  = threadIdx.x;          // 512
  int bh = t >> 5, dh = t & 31;
  int b  = bh >> 2, h = bh & 3;
  int s = src[e], d = dst[e];
  float dv = den[d * 16 + bh];
  float a  = ebuf[e * 16 + bh] / ((dv == 0.f) ? 1.f : dv);
  float msg = feat[((size_t)(b * N_ + s)) * DD + h * DH + dh] * a;
  atomicAdd(&agg[(size_t)d * 512 + bh * DH + dh], msg);
}

// out = relu(agg @ out_w^T + out_b), written to [B,N,H*OUT_L] layout
__global__ void out_fc(const float* __restrict__ agg,
                       const float* __restrict__ ow,
                       const float* __restrict__ ob,
                       float* __restrict__ xout, int N_) {
  int idx = blockIdx.x * blockDim.x + threadIdx.x;
  if (idx >= N_ * 512) return;
  int o  = idx & 31;
  int hh = (idx >> 5) & 3;
  int b  = (idx >> 7) & 3;
  int n  = idx >> 9;
  const float* ag = agg + (size_t)n * 512 + b * 128 + hh * DH;
  const float* w  = ow + o * DH;
  float s = ob[o];
#pragma unroll
  for (int i = 0; i < DH; ++i) s += ag[i] * w[i];
  s = (s > 0.f) ? s : 0.f;
  xout[((size_t)b * N_ + n) * DD + hh * DH + o] = s;
}

// ---------------------------------------------------------------------------
extern "C" void kernel_launch(void* const* d_in, const int* in_sizes, int n_in,
                              void* d_out, int out_size, void* d_ws, size_t ws_size,
                              hipStream_t stream) {
  const int N_ = NNODES, M_ = BB * NNODES;
  const int E_ = in_sizes[1];

  const float* features = (const float*)d_in[0];
  const int*   src      = (const int*)d_in[1];
  const int*   dst      = (const int*)d_in[2];
  const float* edge_w   = (const float*)d_in[3];
  const float* fcw[2]   = {(const float*)d_in[4],  (const float*)d_in[9]};
  const float* alp[2]   = {(const float*)d_in[5],  (const float*)d_in[10]};
  const float* arp[2]   = {(const float*)d_in[6],  (const float*)d_in[11]};
  const float* owp[2]   = {(const float*)d_in[7],  (const float*)d_in[12]};
  const float* obp[2]   = {(const float*)d_in[8],  (const float*)d_in[13]};
  const float* mlp_w    = (const float*)d_in[14];
  const float* mlp_b    = (const float*)d_in[15];

  // workspace carve-up (floats)
  float* ws = (float*)d_ws;
  size_t o = 0;
  float*    feat = ws + o; o += (size_t)M_ * DD;   // 5.12M
  float*    el   = ws + o; o += (size_t)M_ * HH;
  float*    er   = ws + o; o += (size_t)M_ * HH;
  float*    ebuf = ws + o; o += (size_t)E_ * 16;
  unsigned* menc = (unsigned*)(ws + o); o += (size_t)N_ * 16;
  float*    den  = ws + o; o += (size_t)N_ * 16;
  float*    agg  = ws + o; o += (size_t)N_ * 512;
  float*    x1   = ws + o; o += (size_t)M_ * DD;
  float*    x2   = ws + o; o += (size_t)M_ * DD;

  const float* xin[2] = {features, x1};
  float*       xout[2] = {x1, x2};

  dim3 gG((M_ + 63) / 64), bG(256);
  int nEBH = E_ * 16;
  int gE   = (nEBH + 255) / 256;
  int gN   = (N_ * 512 + 255) / 256;
  int gLR  = (M_ * HH + 255) / 256;

  for (int l = 0; l < 2; ++l) {
    gemm_xwT_wmma<<<gG, bG, 0, stream>>>(xin[l], nullptr, fcw[l], nullptr,
                                         feat, M_, 1, 128);
    lr_kernel<<<gLR, 256, 0, stream>>>(feat, alp[l], arp[l], el, er, M_ * HH);
    init_kernel<<<gN, 256, 0, stream>>>(menc, den, agg, N_ * 16);
    edge_score<<<gE, 256, 0, stream>>>(src, dst, edge_w, el, er, ebuf, menc, E_, N_);
    edge_exp<<<gE, 256, 0, stream>>>(dst, ebuf, menc, den, E_);
    edge_agg<<<E_, 512, 0, stream>>>(src, dst, ebuf, den, feat, agg, N_);
    out_fc<<<gN, 256, 0, stream>>>(agg, owp[l], obp[l], xout[l], N_);
  }

  // final MLP over concat(x1,x2): K=256 handled as two 128-chunks
  gemm_xwT_wmma<<<gG, bG, 0, stream>>>(x1, x2, mlp_w, mlp_b,
                                       (float*)d_out, M_, 2, 256);
}
